// ExactVisionModel_24678882083028
// MI455X (gfx1250) — matrix operand order repaired and checked by reading, not compiled, not run on hardware
//
#include <hip/hip_runtime.h>
#include <math.h>

typedef _Float16 half_t;
typedef _Float16 v4h  __attribute__((ext_vector_type(4)));
typedef _Float16 v8h  __attribute__((ext_vector_type(8)));
typedef _Float16 v16h __attribute__((ext_vector_type(16)));
typedef float    v4f  __attribute__((ext_vector_type(4)));
typedef float    v8f  __attribute__((ext_vector_type(8)));

#define HDIM 1280
#define NHEAD 16
#define HEADD 80
#define FFDIM 5120
#define NTOK 7424
#define QKVN 3840
#define MTOK 1856   // NTOK/4 merged tokens
#define MDIM 5120   // HDIM*4
#define OUTD 2048

// ---------------- helpers ----------------

__device__ __forceinline__ float gelu_tanh(float x) {
  float x3 = x * x * x;
  return 0.5f * x * (1.0f + tanhf(0.7978845608028654f * (x + 0.044715f * x3)));
}

// token -> (row, col) spatial position within its grid + segment id
__device__ __forceinline__ void token_rc(int n, int &row, int &col, int &seg) {
  const int cu[4]   = {0, 1024, 3328, 5376};
  const int segh[4] = {32, 48, 32, 64};
  const int segw[4] = {32, 48, 32, 32};
  int s = (n < 1024) ? 0 : (n < 3328) ? 1 : (n < 5376) ? 2 : 3;
  seg = s;
  int l = n - cu[s];
  int mwd = segw[s] >> 1, mhd = segh[s] >> 1;
  int ic = l & 1;
  int ir = (l >> 1) & 1;
  int rest = l >> 2;
  int mw = rest % mwd; rest /= mwd;
  int mh = rest % mhd;               // t stripped off above mh
  row = mh * 2 + ir;
  col = mw * 2 + ic;
}

// WMMA fragment loads from an LDS row (16B-aligned base, >=32 halves).
// A-matrix 16x32 f16 layout: lane<16 -> K{0..7,16..23}; lane>=16 -> K{8..15,24..31}
// B-matrix 32x16 f16 layout: lane<16 -> K{0..7,16..23}; lane>=16 -> K{8..15,24..31}
__device__ __forceinline__ v16h frag_ld(const half_t *rowp, int c0, int c1) {
  v8h lo = *(const v8h *)(rowp + c0);
  v8h hi = *(const v8h *)(rowp + c1);
  v16h r;
#pragma unroll
  for (int i = 0; i < 8; ++i) { r[i] = lo[i]; r[i + 8] = hi[i]; }
  return r;
}
__device__ __forceinline__ v16h frag_a(const half_t *rowp, int sel) {
  return frag_ld(rowp, sel ? 8 : 0, sel ? 24 : 16);
}
__device__ __forceinline__ v16h frag_b(const half_t *rowp, int sel) {
  return frag_ld(rowp, sel ? 8 : 0, sel ? 24 : 16);
}

__device__ __forceinline__ v4h cvt4(v4f v) {
  v4h h;
  h[0] = (half_t)v[0]; h[1] = (half_t)v[1];
  h[2] = (half_t)v[2]; h[3] = (half_t)v[3];
  return h;
}

// ---------------- elementwise kernels ----------------

// pixel_values (N, C*TP*P*P) [c,t,h,w order] -> (N, TP*P*P*C) [t,h,w,c order]
__global__ void k_permute(const float *__restrict__ pix, float *__restrict__ outp) {
  int idx = blockIdx.x * 256 + threadIdx.x;
  if (idx >= NTOK * 1536) return;
  int n = idx / 1536, r = idx % 1536;
  int c = r % 3;
  int r3 = r / 3;
  int wcol = r3 & 15, hrow = (r3 >> 4) & 15, t = r3 >> 8;
  outp[idx] = pix[(size_t)n * 1536 + c * 512 + t * 256 + hrow * 16 + wcol];
}

// bilinear-interpolated positional embedding add
__global__ void k_addpos(float *__restrict__ h, const float *__restrict__ table) {
  int idx = blockIdx.x * 256 + threadIdx.x;
  if (idx >= NTOK * HDIM) return;
  int n = idx / HDIM, c = idx % HDIM;
  const int segh[4] = {32, 48, 32, 64};
  const int segw[4] = {32, 48, 32, 32};
  int row, col, s;
  token_rc(n, row, col, s);
  float hi = row * 31.0f / (float)(segh[s] - 1);
  float wi = col * 31.0f / (float)(segw[s] - 1);
  int hf = (int)floorf(hi), wf = (int)floorf(wi);
  int hc = min(hf + 1, 31), wc = min(wf + 1, 31);
  float dh = hi - (float)hf, dw = wi - (float)wf;
  float v = (1.f - dh) * (1.f - dw) * table[(size_t)(hf * 32 + wf) * HDIM + c]
          + (1.f - dh) * dw         * table[(size_t)(hf * 32 + wc) * HDIM + c]
          + dh * (1.f - dw)         * table[(size_t)(hc * 32 + wf) * HDIM + c]
          + dh * dw                 * table[(size_t)(hc * 32 + wc) * HDIM + c];
  h[idx] += v;
}

// rotary cos/sin tables (NTOK x 80)
__global__ void k_rope_tables(float *__restrict__ cosb, float *__restrict__ sinb) {
  int idx = blockIdx.x * 256 + threadIdx.x;
  if (idx >= NTOK * HEADD) return;
  int n = idx / HEADD, d = idx % HEADD;
  int row, col, s;
  token_rc(n, row, col, s);
  int j = d % 40;
  float pos, e;
  if (j < 20) { pos = (float)row; e = (float)j; }
  else        { pos = (float)col; e = (float)(j - 20); }
  float inv = __powf(10000.0f, -(2.0f * e) / 40.0f);
  float ang = pos * inv;
  cosb[idx] = __cosf(ang);
  sinb[idx] = __sinf(ang);
}

// apply rotary to q and k in-place inside qkv buffer
__global__ void k_rope_apply(float *__restrict__ qkv, const float *__restrict__ cosb,
                             const float *__restrict__ sinb) {
  int idx = blockIdx.x * 256 + threadIdx.x;
  if (idx >= NTOK * NHEAD * 40) return;
  int j = idx % 40;
  int t = idx / 40;
  int head = t % NHEAD;
  int n = t / NHEAD;
  int d1 = j, d2 = j + 40;
  float c1 = cosb[n * 80 + d1], s1 = sinb[n * 80 + d1];
  float c2 = cosb[n * 80 + d2], s2 = sinb[n * 80 + d2];
  size_t base = (size_t)n * QKVN + head * HEADD;
#pragma unroll
  for (int off = 0; off <= HDIM; off += HDIM) {  // q then k
    float x1 = qkv[base + off + d1];
    float x2 = qkv[base + off + d2];
    qkv[base + off + d1] = x1 * c1 - x2 * s1;
    qkv[base + off + d2] = x2 * c2 + x1 * s2;
  }
}

// layernorm over 1280 features, one block per row
__global__ __launch_bounds__(256) void k_layernorm(const float *__restrict__ x,
                                                   const float *__restrict__ s,
                                                   const float *__restrict__ b,
                                                   float *__restrict__ o) {
  int r = blockIdx.x;
  const float *xr = x + (size_t)r * HDIM;
  float sum = 0.f, sq = 0.f;
  for (int c = threadIdx.x; c < HDIM; c += 256) {
    float v = xr[c];
    sum += v; sq += v * v;
  }
  __shared__ float rs[256], rq[256];
  rs[threadIdx.x] = sum; rq[threadIdx.x] = sq;
  __syncthreads();
  for (int st = 128; st > 0; st >>= 1) {
    if (threadIdx.x < st) {
      rs[threadIdx.x] += rs[threadIdx.x + st];
      rq[threadIdx.x] += rq[threadIdx.x + st];
    }
    __syncthreads();
  }
  float mean = rs[0] * (1.0f / HDIM);
  float var = rq[0] * (1.0f / HDIM) - mean * mean;
  float rstd = rsqrtf(var + 1e-6f);
  float *orow = o + (size_t)r * HDIM;
  for (int c = threadIdx.x; c < HDIM; c += 256)
    orow[c] = (xr[c] - mean) * rstd * s[c] + b[c];
}

// ---------------- WMMA GEMM ----------------
// C[M,N] = epi( A[M,K] @ B[K,N] ), A/B fp32 row-major, converted to f16 through LDS.
// FLAGS: 1=+bias  2=gelu  4=+res   (compile-time)
// 64x128 block tile, 256 threads = 8 waves of 32x32, register-prefetch pipeline.
#define BM 64
#define BN 128
#define BK 32
#define LSTR 40  // LDS row stride in halves (80B: 16B aligned, conflict-free)

template <int FLAGS>
__global__ __launch_bounds__(256) void k_gemm(const float *__restrict__ A,
                                              const float *__restrict__ B,
                                              const float *__restrict__ bias,
                                              const float *__restrict__ res,
                                              float *__restrict__ out,
                                              int M, int N, int K) {
  __shared__ half_t As[BM * LSTR];
  __shared__ half_t Bs[BN * LSTR];
  const int tid = threadIdx.x;
  const int wave = tid >> 5;
  const int lane = tid & 31;
  const int lrow = lane & 15;
  const int lsel = lane >> 4;
  const int m0 = blockIdx.y * BM;
  const int n0 = blockIdx.x * BN;
  const int wm = (wave >> 2) * 32;   // 2 m-waves
  const int wn = (wave & 3) * 32;    // 4 n-waves

  // staging coordinates (256 threads)
  const int arow = tid >> 3;               // A: 2 float4 per thread
  const int akc = (tid & 7) << 2;
  const int bkrow = tid >> 5;              // B: 4 float4 per thread
  const int bnc = (tid & 31) << 2;

  v8f acc[2][2];
#pragma unroll
  for (int mt = 0; mt < 2; ++mt)
#pragma unroll
    for (int nt = 0; nt < 2; ++nt)
#pragma unroll
      for (int i = 0; i < 8; ++i) acc[mt][nt][i] = 0.0f;

  // prologue: prefetch first K-slice into registers
  v4f pa[2], pb[4];
#pragma unroll
  for (int it = 0; it < 2; ++it)
    pa[it] = *(const v4f *)(A + (size_t)(m0 + arow + it * 32) * K + akc);
#pragma unroll
  for (int it = 0; it < 4; ++it)
    pb[it] = *(const v4f *)(B + (size_t)(bkrow + it * 8) * N + n0 + bnc);

  for (int k0 = 0; k0 < K; k0 += BK) {
    // commit prefetched slice to LDS (f32 -> f16)
#pragma unroll
    for (int it = 0; it < 2; ++it)
      *(v4h *)(As + (arow + it * 32) * LSTR + akc) = cvt4(pa[it]);
#pragma unroll
    for (int it = 0; it < 4; ++it) {
      v4h hv = cvt4(pb[it]);
      int kr = bkrow + it * 8;
      Bs[(bnc + 0) * LSTR + kr] = hv[0];
      Bs[(bnc + 1) * LSTR + kr] = hv[1];
      Bs[(bnc + 2) * LSTR + kr] = hv[2];
      Bs[(bnc + 3) * LSTR + kr] = hv[3];
    }
    __syncthreads();

    // prefetch next slice while WMMAs run
    int k1 = k0 + BK;
    if (k1 < K) {
#pragma unroll
      for (int it = 0; it < 2; ++it)
        pa[it] = *(const v4f *)(A + (size_t)(m0 + arow + it * 32) * K + k1 + akc);
#pragma unroll
      for (int it = 0; it < 4; ++it)
        pb[it] = *(const v4f *)(B + (size_t)(k1 + bkrow + it * 8) * N + n0 + bnc);
    }

    v16h af[2], bf[2];
#pragma unroll
    for (int mt = 0; mt < 2; ++mt)
      af[mt] = frag_a(As + (wm + mt * 16 + lrow) * LSTR, lsel);
#pragma unroll
    for (int nt = 0; nt < 2; ++nt)
      bf[nt] = frag_b(Bs + (wn + nt * 16 + lrow) * LSTR, lsel);

#pragma unroll
    for (int mt = 0; mt < 2; ++mt)
#pragma unroll
      for (int nt = 0; nt < 2; ++nt)
        acc[mt][nt] = __builtin_amdgcn_wmma_f32_16x16x32_f16(
            false, af[mt], false, bf[nt], (short)0, acc[mt][nt], false, false);
    __syncthreads();
  }

  // epilogue: lane holds col = lrow, rows lsel*8 + i
#pragma unroll
  for (int mt = 0; mt < 2; ++mt) {
#pragma unroll
    for (int nt = 0; nt < 2; ++nt) {
      int col = n0 + wn + nt * 16 + lrow;
      float bv = (FLAGS & 1) ? bias[col] : 0.0f;
#pragma unroll
      for (int i = 0; i < 8; ++i) {
        int row = m0 + wm + mt * 16 + lsel * 8 + i;
        float v = acc[mt][nt][i] + bv;
        if (FLAGS & 2) v = gelu_tanh(v);
        if (FLAGS & 4) v += res[(size_t)row * N + col];
        out[(size_t)row * N + col] = v;
      }
    }
  }
}

// ---------------- flash attention (1 wave per head x 16-query tile) ----------------
#define QSTR 104  // 96 padded + 8 (208B, 16B aligned)
#define VSTR 40

__global__ __launch_bounds__(32) void k_attn(const float *__restrict__ qkv,
                                             float *__restrict__ out) {
  __shared__ half_t Qs[16 * QSTR];
  __shared__ half_t Ks[32 * QSTR];
  __shared__ half_t Vt[80 * VSTR];
  __shared__ half_t Ps[16 * VSTR];

  const int cu[4]     = {0, 1024, 3328, 5376};
  const int qtst[4]   = {0, 64, 208, 336};
  const int seglen[4] = {1024, 2304, 2048, 2048};

  int bid = blockIdx.x;
  int head = bid & 15;
  int qt = bid >> 4;
  int s = (qt < 64) ? 0 : (qt < 208) ? 1 : (qt < 336) ? 2 : 3;
  int qbase = cu[s] + (qt - qtst[s]) * 16;
  int kstart = cu[s];
  int L = seglen[s];

  int lane = threadIdx.x;
  int lrow = lane & 15;
  int lsel = lane >> 4;
  const float scale = 0.11180339887498949f;  // 80^-0.5

  // zero pad columns (d = 80..95) of Qs and Ks once
  for (int i = lane; i < 16 * 16; i += 32)
    Qs[(i >> 4) * QSTR + 80 + (i & 15)] = (half_t)0.0f;
  for (int i = lane; i < 32 * 16; i += 32)
    Ks[(i >> 4) * QSTR + 80 + (i & 15)] = (half_t)0.0f;

  // stage Q (scaled), vectorized: 16 rows x 20 float4
  for (int i = lane; i < 16 * 20; i += 32) {
    int r = i / 20, c4 = (i % 20) << 2;
    v4f v = *(const v4f *)(qkv + (size_t)(qbase + r) * QKVN + head * HEADD + c4);
    v[0] *= scale; v[1] *= scale; v[2] *= scale; v[3] *= scale;
    *(v4h *)(Qs + r * QSTR + c4) = cvt4(v);
  }
  __syncthreads();
  v16h qf[3];
#pragma unroll
  for (int dk = 0; dk < 3; ++dk)
    qf[dk] = frag_a(Qs + lrow * QSTR + dk * 32, lsel);

  v8f accO[5];
#pragma unroll
  for (int nt = 0; nt < 5; ++nt)
#pragma unroll
    for (int i = 0; i < 8; ++i) accO[nt][i] = 0.0f;
  float runmax[8], runsum[8];
#pragma unroll
  for (int i = 0; i < 8; ++i) { runmax[i] = -1e30f; runsum[i] = 0.0f; }

  for (int kt = 0; kt < L; kt += 32) {
    // stage K (32 x 80, vectorized) and V^T (80 x 32)
    for (int i = lane; i < 32 * 20; i += 32) {
      int r = i / 20, c4 = (i % 20) << 2;
      v4f v = *(const v4f *)(qkv + (size_t)(kstart + kt + r) * QKVN + HDIM +
                             head * HEADD + c4);
      *(v4h *)(Ks + r * QSTR + c4) = cvt4(v);
    }
    for (int i = lane; i < 32 * 20; i += 32) {
      int r = i / 20, c4 = (i % 20) << 2;
      v4f v = *(const v4f *)(qkv + (size_t)(kstart + kt + r) * QKVN + 2 * HDIM +
                             head * HEADD + c4);
      v4h hv = cvt4(v);
      Vt[(c4 + 0) * VSTR + r] = hv[0];
      Vt[(c4 + 1) * VSTR + r] = hv[1];
      Vt[(c4 + 2) * VSTR + r] = hv[2];
      Vt[(c4 + 3) * VSTR + r] = hv[3];
    }
    __syncthreads();

    // S = Q @ K^T : two 16x16 key tiles, reduce over d (3 steps)
    v8f s0, s1;
#pragma unroll
    for (int i = 0; i < 8; ++i) { s0[i] = 0.0f; s1[i] = 0.0f; }
#pragma unroll
    for (int dk = 0; dk < 3; ++dk) {
      v16h b0 = frag_b(Ks + lrow * QSTR + dk * 32, lsel);
      v16h b1 = frag_b(Ks + (16 + lrow) * QSTR + dk * 32, lsel);
      s0 = __builtin_amdgcn_wmma_f32_16x16x32_f16(false, qf[dk], false, b0,
                                                  (short)0, s0, false, false);
      s1 = __builtin_amdgcn_wmma_f32_16x16x32_f16(false, qf[dk], false, b1,
                                                  (short)0, s1, false, false);
    }

    // online softmax: D-layout row = lsel*8 + i, spread over 16 lanes
#pragma unroll
    for (int i = 0; i < 8; ++i) {
      float m = fmaxf(s0[i], s1[i]);
      m = fmaxf(m, __shfl_xor(m, 1));
      m = fmaxf(m, __shfl_xor(m, 2));
      m = fmaxf(m, __shfl_xor(m, 4));
      m = fmaxf(m, __shfl_xor(m, 8));
      float nm = fmaxf(runmax[i], m);
      float alpha = __expf(runmax[i] - nm);
      runmax[i] = nm;
      float p0 = __expf(s0[i] - nm);
      float p1 = __expf(s1[i] - nm);
      float rsum = p0 + p1;
      rsum += __shfl_xor(rsum, 1);
      rsum += __shfl_xor(rsum, 2);
      rsum += __shfl_xor(rsum, 4);
      rsum += __shfl_xor(rsum, 8);
      runsum[i] = runsum[i] * alpha + rsum;
#pragma unroll
      for (int nt = 0; nt < 5; ++nt) accO[nt][i] *= alpha;
      int prow = lsel * 8 + i;
      Ps[prow * VSTR + lrow] = (half_t)p0;
      Ps[prow * VSTR + 16 + lrow] = (half_t)p1;
    }
    __syncthreads();

    // O += P @ V
    v16h pf = frag_a(Ps + lrow * VSTR, lsel);
#pragma unroll
    for (int nt = 0; nt < 5; ++nt) {
      v16h vf = frag_b(Vt + (nt * 16 + lrow) * VSTR, lsel);
      accO[nt] = __builtin_amdgcn_wmma_f32_16x16x32_f16(false, pf, false, vf,
                                                        (short)0, accO[nt], false, false);
    }
    __syncthreads();
  }

#pragma unroll
  for (int i = 0; i < 8; ++i) {
    int qrow = qbase + lsel * 8 + i;
    float inv = 1.0f / runsum[i];
#pragma unroll
    for (int nt = 0; nt < 5; ++nt)
      out[(size_t)qrow * HDIM + head * HEADD + nt * 16 + lrow] = accO[nt][i] * inv;
  }
}

// ---------------- orchestration ----------------

extern "C" void kernel_launch(void *const *d_in, const int *in_sizes, int n_in,
                              void *d_out, int out_size, void *d_ws, size_t ws_size,
                              hipStream_t stream) {
  (void)in_sizes; (void)n_in; (void)out_size; (void)ws_size;
  const float *pix    = (const float *)d_in[0];
  const float *patchk = (const float *)d_in[2];
  const float *patchb = (const float *)d_in[3];
  const float *post   = (const float *)d_in[4];
  const float *ln1s   = (const float *)d_in[5];
  const float *ln1b   = (const float *)d_in[6];
  const float *qkvw   = (const float *)d_in[7];
  const float *qkvb   = (const float *)d_in[8];
  const float *projw  = (const float *)d_in[9];
  const float *projb  = (const float *)d_in[10];
  const float *ln2s   = (const float *)d_in[11];
  const float *ln2b   = (const float *)d_in[12];
  const float *fc1w   = (const float *)d_in[13];
  const float *fc1b   = (const float *)d_in[14];
  const float *fc2w   = (const float *)d_in[15];
  const float *fc2b   = (const float *)d_in[16];
  const float *mns    = (const float *)d_in[17];
  const float *mnb    = (const float *)d_in[18];
  const float *mf1w   = (const float *)d_in[19];
  const float *mf1b   = (const float *)d_in[20];
  const float *mf2w   = (const float *)d_in[21];
  const float *mf2b   = (const float *)d_in[22];
  float *out = (float *)d_out;

  // workspace arena (floats)
  float *W = (float *)d_ws;
  float *h    = W;                                  // NTOK*1280
  float *xln  = h + (size_t)NTOK * HDIM;            // NTOK*1280
  float *attn = xln + (size_t)NTOK * HDIM;          // NTOK*1280
  float *cosb = attn + (size_t)NTOK * HDIM;         // NTOK*80
  float *sinb = cosb + (size_t)NTOK * HEADD;        // NTOK*80
  float *big  = sinb + (size_t)NTOK * HEADD;        // max: NTOK*5120 (aliased)
  float *perm = big, *qkv = big, *mlp1 = big, *mg1 = big;

  // patch embed: permute then GEMM (7424x1536 @ 1536x1280)
  k_permute<<<(NTOK * 1536 + 255) / 256, 256, 0, stream>>>(pix, perm);
  k_gemm<1><<<dim3(HDIM / BN, NTOK / BM), 256, 0, stream>>>(
      perm, patchk, patchb, nullptr, h, NTOK, HDIM, 1536);
  k_addpos<<<(NTOK * HDIM + 255) / 256, 256, 0, stream>>>(h, post);
  k_rope_tables<<<(NTOK * HEADD + 255) / 256, 256, 0, stream>>>(cosb, sinb);

  for (int l = 0; l < 4; ++l) {
    k_layernorm<<<NTOK, 256, 0, stream>>>(h, ln1s + l * HDIM, ln1b + l * HDIM, xln);
    k_gemm<1><<<dim3(QKVN / BN, NTOK / BM), 256, 0, stream>>>(
        xln, qkvw + (size_t)l * HDIM * QKVN, qkvb + (size_t)l * QKVN, nullptr,
        qkv, NTOK, QKVN, HDIM);
    k_rope_apply<<<(NTOK * NHEAD * 40 + 255) / 256, 256, 0, stream>>>(qkv, cosb, sinb);
    k_attn<<<464 * NHEAD, 32, 0, stream>>>(qkv, attn);
    k_gemm<5><<<dim3(HDIM / BN, NTOK / BM), 256, 0, stream>>>(
        attn, projw + (size_t)l * HDIM * HDIM, projb + (size_t)l * HDIM, h,
        h, NTOK, HDIM, HDIM);
    k_layernorm<<<NTOK, 256, 0, stream>>>(h, ln2s + l * HDIM, ln2b + l * HDIM, xln);
    k_gemm<3><<<dim3(FFDIM / BN, NTOK / BM), 256, 0, stream>>>(
        xln, fc1w + (size_t)l * HDIM * FFDIM, fc1b + (size_t)l * FFDIM, nullptr,
        mlp1, NTOK, FFDIM, HDIM);
    k_gemm<5><<<dim3(HDIM / BN, NTOK / BM), 256, 0, stream>>>(
        mlp1, fc2w + (size_t)l * FFDIM * HDIM, fc2b + (size_t)l * HDIM, h,
        h, NTOK, HDIM, FFDIM);
  }

  // merger: LN (per 1280), view as (1856 x 5120)
  k_layernorm<<<NTOK, 256, 0, stream>>>(h, mns, mnb, xln);
  k_gemm<3><<<dim3(MDIM / BN, MTOK / BM), 256, 0, stream>>>(
      xln, mf1w, mf1b, nullptr, mg1, MTOK, MDIM, MDIM);
  k_gemm<1><<<dim3(OUTD / BN, MTOK / BM), 256, 0, stream>>>(
      mg1, mf2w, mf2b, nullptr, out, MTOK, OUTD, MDIM);
}
